// ContextualLSTMTransformerFlexible_27831388078616
// MI455X (gfx1250) — matrix operand order repaired and checked by reading, not compiled, not run on hardware
//
#include <hip/hip_runtime.h>
#include <hip/hip_bf16.h>

// ---------------------------------------------------------------------------
// Model dims
// ---------------------------------------------------------------------------
#define SEQ_LEN   4096
#define WINLEN    32
#define FEAT      32
#define NWIN      128          // NW = SEQ_LEN / WIN
#define LSTM_U    64
#define NHEADS    4
#define EMB       128
#define KCTX      4            // 2*MAXN
#define OUTD      5
#define BATCH     64
#define NWINDOWS  8192         // BATCH * NWIN

typedef __bf16 bf16_t;
typedef __attribute__((ext_vector_type(16))) __bf16 v16bf;
typedef __attribute__((ext_vector_type(8)))  float  v8f;
typedef __attribute__((ext_vector_type(4)))  unsigned int u32x4;
typedef __attribute__((ext_vector_type(4)))  int i32x4;
typedef __attribute__((ext_vector_type(8)))  int i32x8;

#ifndef __has_builtin
#define __has_builtin(x) 0
#endif

#if __has_builtin(__builtin_amdgcn_tensor_load_to_lds) && \
    __has_builtin(__builtin_amdgcn_s_wait_tensorcnt)
#define USE_TDM 1
#else
#define USE_TDM 0
#endif

// ---------------------------------------------------------------------------
// Fast transcendental helpers (single TRANS32 ops; co-execute with WMMA)
// ---------------------------------------------------------------------------
__device__ __forceinline__ float fast_rcp(float x) {
#if __has_builtin(__builtin_amdgcn_rcpf)
  return __builtin_amdgcn_rcpf(x);
#else
  return 1.f / x;
#endif
}
__device__ __forceinline__ float fast_rsqrt(float x) {
#if __has_builtin(__builtin_amdgcn_rsqf)
  return __builtin_amdgcn_rsqf(x);
#else
  return rsqrtf(x);
#endif
}
__device__ __forceinline__ float fast_tanh(float x) {
#if __has_builtin(__builtin_amdgcn_tanhf)
  return __builtin_amdgcn_tanhf(x);
#else
  return tanhf(x);
#endif
}
__device__ __forceinline__ float sigm(float x) {
  return fast_rcp(1.f + __expf(-x));
}

// ---------------------------------------------------------------------------
// WMMA helpers — layouts per CDNA5 ISA 7.12.2 (wave32)
// ---------------------------------------------------------------------------
__device__ __forceinline__ v8f wmma_bf(v16bf a, v16bf b, v8f c) {
  // v_wmma_f32_16x16x32_bf16  D = A(16x32) * B(32x16) + C(16x16 f32)
  return __builtin_amdgcn_wmma_f32_16x16x32_bf16(
      /*neg_a=*/false, a, /*neg_b=*/false, b,
      /*c_mod=*/(short)0, c, /*reuse_a=*/false, /*reuse_b=*/false);
}

__device__ __forceinline__ int lane_id() { return (int)(threadIdx.x & 31u); }
__device__ __forceinline__ int c_col()   { return (int)(threadIdx.x & 15u); }
__device__ __forceinline__ int c_row0()  { return (int)(((threadIdx.x & 31u) >> 4) << 3); }

// A matrix 16x32 bf16, src row-major [16][ld]:
//   lanes 0-15: M=lane, e0..7 -> K0..7, e8..15 -> K16..23
//   lanes16-31: M=lane-16, e0..7 -> K8..15, e8..15 -> K24..31
__device__ __forceinline__ v16bf load_A16(const bf16_t* src, int ld) {
  const int lane = lane_id();
  const int m  = lane & 15;
  const int kb = (lane >> 4) << 3;
  v16bf a;
#pragma unroll
  for (int e = 0; e < 16; ++e) {
    const int k = kb + (e < 8 ? e : e + 8);
    a[e] = src[m * ld + k];
  }
  return a;
}

// B matrix 32x16 bf16 (K x N), src row-major [32][ld]:
//   lanes 0-15: N=lane, e0..7 -> K0..7, e8..15 -> K16..23 ;  lanes 16-31: N=lane-16, e0..7 -> K8..15, e8..15 -> K24..31
__device__ __forceinline__ v16bf load_B16(const bf16_t* src, int ld) {
  const int lane = lane_id();
  const int n  = lane & 15;
  const int kb = (lane >> 4) << 3;
  v16bf b;
#pragma unroll
  for (int e = 0; e < 16; ++e) b[e] = src[(kb + (e < 8 ? e : e + 8)) * ld + n];
  return b;
}

// C/D 16x16 f32: lane col = lane&15, rows v + 8*(lane>=16)
__device__ __forceinline__ void store_C_f32(float* dst, int ld, v8f c) {
  const int col = c_col(), r0 = c_row0();
#pragma unroll
  for (int v = 0; v < 8; ++v) dst[(r0 + v) * ld + col] = c[v];
}
__device__ __forceinline__ void store_C_bf16(bf16_t* dst, int ld, v8f c) {
  const int col = c_col(), r0 = c_row0();
#pragma unroll
  for (int v = 0; v < 8; ++v) dst[(r0 + v) * ld + col] = (bf16_t)c[v];
}
// transposed store: element (row,col) -> dst[col*ld + row]
__device__ __forceinline__ void store_C_bf16_T(bf16_t* dst, int ld, v8f c) {
  const int col = c_col(), r0 = c_row0();
#pragma unroll
  for (int v = 0; v < 8; ++v) dst[col * ld + (r0 + v)] = (bf16_t)c[v];
}

// ---------------------------------------------------------------------------
// K0: batched transpose f32 -> bf16   dst[b][c][r] = src[b][r][c]
// ---------------------------------------------------------------------------
__global__ void bt_kernel(const float* __restrict__ src, bf16_t* __restrict__ dst,
                          int batch, int rows, int cols) {
  const long long total = (long long)batch * rows * cols;
  for (long long i = (long long)blockIdx.x * blockDim.x + threadIdx.x; i < total;
       i += (long long)gridDim.x * blockDim.x) {
    const int c = (int)(i % cols);
    const long long t = i / cols;
    const int r = (int)(t % rows);
    const int b = (int)(t / rows);
    dst[((long long)b * cols + c) * rows + r] = (bf16_t)src[i];
  }
}

// ---------------------------------------------------------------------------
// K1: bidirectional LSTM.  Block = 256 threads (8 waves) = 16 windows,
// blockIdx.y = direction.  Each wave owns 2 gate N-tiles; B operands live in
// VGPRs for the whole 32-step scan.  3 WMMAs per tile per step.
// ---------------------------------------------------------------------------
__global__ __launch_bounds__(256) void lstm_kernel(
    const float* __restrict__ x,
    const bf16_t* __restrict__ WihT_f, const bf16_t* __restrict__ WhhT_f,
    const float* __restrict__ bih_f,   const float* __restrict__ bhh_f,
    const bf16_t* __restrict__ WihT_b, const bf16_t* __restrict__ WhhT_b,
    const float* __restrict__ bih_b,   const float* __restrict__ bhh_b,
    bf16_t* __restrict__ h_all) {
  const int dir   = blockIdx.y;
  const int wbase = blockIdx.x * 16;
  const int wv    = threadIdx.x >> 5;

  __shared__ bf16_t sX[16 * FEAT];      // x_t tile, row-major [win][feat]
  __shared__ bf16_t sH[16 * LSTM_U];    // h state (A-readable row-major)
  __shared__ float  sC[16 * LSTM_U];    // c state
  __shared__ float  sG[16 * 256];       // gate pre-activations

  const bf16_t* WihT = dir ? WihT_b : WihT_f;   // [32][256]
  const bf16_t* WhhT = dir ? WhhT_b : WhhT_f;   // [64][256]
  const float*  bih  = dir ? bih_b  : bih_f;
  const float*  bhh  = dir ? bhh_b  : bhh_f;

  v16bf Bi[2], Bh0[2], Bh1[2];
  float biasr[2];
#pragma unroll
  for (int j = 0; j < 2; ++j) {
    const int tn = wv * 2 + j;                  // gate N-tile 0..15
    Bi[j]  = load_B16(WihT + tn * 16, 256);
    Bh0[j] = load_B16(WhhT + tn * 16, 256);
    Bh1[j] = load_B16(WhhT + 32 * 256 + tn * 16, 256);
    const int ncol = tn * 16 + c_col();
    biasr[j] = bih[ncol] + bhh[ncol];
  }

  for (int i = threadIdx.x; i < 16 * LSTM_U; i += 256) {
    sH[i] = (bf16_t)0.f;
    sC[i] = 0.f;
  }
  __syncthreads();

  for (int t = 0; t < WINLEN; ++t) {
    const int ts = dir ? (WINLEN - 1 - t) : t;
    // stage x_t for 16 windows (and prefetch next step)
    for (int i = threadIdx.x; i < 16 * FEAT; i += 256) {
      const int w = i >> 5, f = i & 31;
      const int gw = wbase + w;
      const int b  = gw >> 7, cw = gw & 127;
      const size_t base = ((size_t)b * SEQ_LEN + (size_t)cw * WINLEN + ts) * FEAT + f;
      sX[i] = (bf16_t)x[base];
      const int tn2 = dir ? (ts - 1) : (ts + 1);
      if (tn2 >= 0 && tn2 < WINLEN)
        __builtin_prefetch(&x[((size_t)b * SEQ_LEN + (size_t)cw * WINLEN + tn2) * FEAT + f], 0, 1);
    }
    __syncthreads();

    const v16bf Ax  = load_A16(sX, FEAT);
    const v16bf Ah0 = load_A16(sH, LSTM_U);
    const v16bf Ah1 = load_A16(sH + 32, LSTM_U);
#pragma unroll
    for (int j = 0; j < 2; ++j) {
      v8f c = {};
      c = wmma_bf(Ax,  Bi[j],  c);
      c = wmma_bf(Ah0, Bh0[j], c);
      c = wmma_bf(Ah1, Bh1[j], c);
#pragma unroll
      for (int v = 0; v < 8; ++v) c[v] += biasr[j];
      store_C_f32(sG + (wv * 2 + j) * 16, 256, c);
    }
    __syncthreads();

    // elementwise cell update: 16 windows x 64 units
    for (int i = threadIdx.x; i < 16 * LSTM_U; i += 256) {
      const int w = i >> 6, u = i & 63;
      const float gi = sG[w * 256 + u];
      const float gf = sG[w * 256 + 64 + u];
      const float gg = sG[w * 256 + 128 + u];
      const float go = sG[w * 256 + 192 + u];
      const float cn = sigm(gf) * sC[i] + sigm(gi) * fast_tanh(gg);
      const float hn = sigm(go) * fast_tanh(cn);
      sC[i] = cn;
      sH[i] = (bf16_t)hn;
      const int gw = wbase + w;
      h_all[((size_t)gw * WINLEN + ts) * EMB + dir * LSTM_U + u] = (bf16_t)hn;
    }
    __syncthreads();
  }
}

// ---------------------------------------------------------------------------
// K2: per-window self-attention + LN + proj + token-mean -> xe.
// One wave per 32-thread block; all GEMMs are WMMA (S=QK^T and P*V are exact
// 16x16x32 shapes since Dh = WIN = 32).  The 8KB h-window tile is fetched via
// the Tensor Data Mover when available (tensor_load_to_lds + s_wait_tensorcnt).
// ---------------------------------------------------------------------------
__global__ __launch_bounds__(32) void attn_kernel(
    const bf16_t* __restrict__ h_all,
    const bf16_t* __restrict__ WqkvT,   // [128][384]
    const float*  __restrict__ sa_bqkv,
    const bf16_t* __restrict__ WoT,     // [128][128]
    const float*  __restrict__ sa_bo,
    const float*  __restrict__ sa_gamma, const float* __restrict__ sa_beta,
    const bf16_t* __restrict__ projWT,  // [128][128]
    const float*  __restrict__ proj_b,
    float* __restrict__ xe_f, bf16_t* __restrict__ xe_bf) {
  const int w = blockIdx.x;                 // global window (b*NWIN + n)
  __shared__ bf16_t sHW[WINLEN * EMB];      // h window (residual + A operand)
  __shared__ bf16_t sQ[WINLEN * 32];
  __shared__ bf16_t sKt[32 * WINLEN];       // K^T  [d][tok]
  __shared__ bf16_t sV[WINLEN * 32];
  __shared__ bf16_t sP[WINLEN * 32];
  __shared__ bf16_t sAT[WINLEN * EMB];      // attn / normalized h
  __shared__ float  sY[WINLEN * EMB];       // f32 scratch (S, out, proj)

#if USE_TDM
  {
    // TDM descriptor (ISA 8.3/8.4): one 1-D tile of 1024 x 8-byte elements
    // (the contiguous 32x128 bf16 window) DMA'd global -> LDS.
    const unsigned long long ga =
        (unsigned long long)(size_t)(h_all + (size_t)w * (WINLEN * EMB));
    const unsigned int lds = (unsigned int)(size_t)(&sHW[0]);   // LDS byte addr
    u32x4 g0;
    g0[0] = 1u;                                                 // count=1 descriptor
    g0[1] = lds;                                                // lds_addr
    g0[2] = (unsigned int)(ga & 0xFFFFFFFFull);                 // global_addr[31:0]
    g0[3] = (unsigned int)((ga >> 32) & 0x1FFFFFFull) | (2u << 30); // [56:32]|type=2
    i32x8 g1;
    g1[0] = (3 << 16);               // workgroup_mask=0, data_size=8B
    g1[1] = (int)(1024u << 16);      // tensor_dim0 = 1024 (bits 79:48)
    g1[2] = (int)(1u << 16);         // tensor_dim1 = 1    (bits 111:80)
    g1[3] = (int)(1024u << 16);      // tile_dim0 = 1024   (bits 127:112)
    g1[4] = 1;                       // tile_dim1 = 1, tile_dim2 = 0
    g1[5] = 1024;                    // tensor_dim0_stride = 1024
    g1[6] = 0;
    g1[7] = 0;
    i32x4 z4 = {0, 0, 0, 0};
#if __clang_major__ >= 23
    i32x8 z8 = {};
    __builtin_amdgcn_tensor_load_to_lds(g0, g1, z4, z4, z8, 0);
#else
    __builtin_amdgcn_tensor_load_to_lds(g0, g1, z4, z4, 0);
#endif
    __builtin_amdgcn_s_wait_tensorcnt(0);
  }
#else
  for (int i = threadIdx.x; i < WINLEN * EMB; i += 32)
    sHW[i] = h_all[(size_t)w * WINLEN * EMB + i];
#endif
  __syncthreads();

  const float scale = 0.17677669529663687f;   // 1/sqrt(32)

  for (int h = 0; h < NHEADS; ++h) {
    // Q/K/V projections for this head: (32x128)@(128x32)
#pragma unroll
    for (int tgt = 0; tgt < 3; ++tgt) {
      const int colbase = tgt * EMB + h * 32;
#pragma unroll
      for (int mi = 0; mi < 2; ++mi)
#pragma unroll
        for (int ni = 0; ni < 2; ++ni) {
          v8f c = {};
#pragma unroll
          for (int kc = 0; kc < 4; ++kc)
            c = wmma_bf(load_A16(sHW + mi * 16 * EMB + kc * 32, EMB),
                        load_B16(WqkvT + (kc * 32) * 384 + colbase + ni * 16, 384), c);
          const float bias = sa_bqkv[colbase + ni * 16 + c_col()];
#pragma unroll
          for (int v = 0; v < 8; ++v) c[v] += bias;
          if (tgt == 0)      store_C_bf16  (sQ  + mi * 16 * 32 + ni * 16, 32, c);
          else if (tgt == 1) store_C_bf16_T(sKt + (ni * 16) * 32 + mi * 16, 32, c);
          else               store_C_bf16  (sV  + mi * 16 * 32 + ni * 16, 32, c);
        }
    }
    __syncthreads();

    // S = Q K^T * scale
#pragma unroll
    for (int mi = 0; mi < 2; ++mi)
#pragma unroll
      for (int ni = 0; ni < 2; ++ni) {
        v8f c = {};
        c = wmma_bf(load_A16(sQ + mi * 16 * 32, 32), load_B16(sKt + ni * 16, 32), c);
#pragma unroll
        for (int v = 0; v < 8; ++v) c[v] *= scale;
        store_C_f32(sY + mi * 16 * 32 + ni * 16, 32, c);
      }
    __syncthreads();

    // row softmax (lane = row)
    {
      const int r = lane_id();
      float mx = -1e30f;
      for (int k = 0; k < 32; ++k) mx = fmaxf(mx, sY[r * 32 + k]);
      float sum = 0.f;
      for (int k = 0; k < 32; ++k) sum += __expf(sY[r * 32 + k] - mx);
      const float inv = fast_rcp(sum);
      for (int k = 0; k < 32; ++k)
        sP[r * 32 + k] = (bf16_t)(__expf(sY[r * 32 + k] - mx) * inv);
    }
    __syncthreads();

    // attn_h = P V
#pragma unroll
    for (int mi = 0; mi < 2; ++mi)
#pragma unroll
      for (int ni = 0; ni < 2; ++ni) {
        v8f c = {};
        c = wmma_bf(load_A16(sP + mi * 16 * 32, 32), load_B16(sV + ni * 16, 32), c);
        store_C_bf16(sAT + mi * 16 * EMB + h * 32 + ni * 16, EMB, c);
      }
    __syncthreads();
  }

  // out = attn @ Wo^T + bo + residual
#pragma unroll
  for (int mi = 0; mi < 2; ++mi)
    for (int ni = 0; ni < 8; ++ni) {
      v8f c = {};
#pragma unroll
      for (int kc = 0; kc < 4; ++kc)
        c = wmma_bf(load_A16(sAT + mi * 16 * EMB + kc * 32, EMB),
                    load_B16(WoT + (kc * 32) * EMB + ni * 16, EMB), c);
      const int col = ni * 16 + c_col();
#pragma unroll
      for (int v = 0; v < 8; ++v) {
        const int row = mi * 16 + c_row0() + v;
        c[v] += sa_bo[col] + (float)sHW[row * EMB + col];
      }
      store_C_f32(sY + mi * 16 * EMB + ni * 16, EMB, c);
    }
  __syncthreads();

  // LayerNorm per token -> sAT (bf16)
  {
    const int r = lane_id();
    float m = 0.f;
    for (int e = 0; e < EMB; ++e) m += sY[r * EMB + e];
    m *= (1.f / EMB);
    float var = 0.f;
    for (int e = 0; e < EMB; ++e) { const float d = sY[r * EMB + e] - m; var += d * d; }
    var *= (1.f / EMB);
    const float rinv = fast_rsqrt(var + 1e-5f);
    for (int e = 0; e < EMB; ++e)
      sAT[r * EMB + e] = (bf16_t)((sY[r * EMB + e] - m) * rinv * sa_gamma[e] + sa_beta[e]);
  }
  __syncthreads();

  // proj: h'' = h' @ proj_W^T + proj_b
#pragma unroll
  for (int mi = 0; mi < 2; ++mi)
    for (int ni = 0; ni < 8; ++ni) {
      v8f c = {};
#pragma unroll
      for (int kc = 0; kc < 4; ++kc)
        c = wmma_bf(load_A16(sAT + mi * 16 * EMB + kc * 32, EMB),
                    load_B16(projWT + (kc * 32) * EMB + ni * 16, EMB), c);
      const float bias = proj_b[ni * 16 + c_col()];
#pragma unroll
      for (int v = 0; v < 8; ++v) c[v] += bias;
      store_C_f32(sY + mi * 16 * EMB + ni * 16, EMB, c);
    }
  __syncthreads();

  // token-mean -> xe
  {
    const int l = lane_id();
    for (int cc = l; cc < EMB; cc += 32) {
      float s = 0.f;
      for (int r = 0; r < WINLEN; ++r) s += sY[r * EMB + cc];
      s *= (1.f / WINLEN);
      const size_t o = (size_t)w * EMB + cc;
      xe_f[o]  = s;
      xe_bf[o] = (bf16_t)s;
    }
  }
}

// ---------------------------------------------------------------------------
// K3: per-window cross-attention (per-window weights) + LN -> seq (bf16).
// Block = 128 threads (4 waves), handles window n for a 16-batch chunk.
// ---------------------------------------------------------------------------
__global__ __launch_bounds__(128) void cross_kernel(
    const bf16_t* __restrict__ xe_bf, const float* __restrict__ xe_f,
    const bf16_t* __restrict__ caWqkvT,  // [NWIN][128][384]
    const float*  __restrict__ ca_bqkv,  // [NWIN][384]
    const bf16_t* __restrict__ caWoT,    // [NWIN][128][128]
    const float*  __restrict__ ca_bo,    // [NWIN][128]
    const float*  __restrict__ ca_gamma, const float* __restrict__ ca_beta,
    const int* __restrict__ ctx_idx, const unsigned char* __restrict__ ctx_mask,
    bf16_t* __restrict__ seq_bf) {
  const int n     = blockIdx.x;
  const int bbase = blockIdx.y * 16;
  const int wv    = threadIdx.x >> 5;

  __shared__ bf16_t sQ[16 * EMB];
  __shared__ bf16_t sK[KCTX * 16 * EMB];
  __shared__ bf16_t sV[KCTX * 16 * EMB];
  __shared__ bf16_t sO[16 * EMB];
  __shared__ float  sY[16 * EMB];

  const bf16_t* WT = caWqkvT + (size_t)n * EMB * 384;
  const int ldXe = NWIN * EMB;   // row stride over batch index

  // 9 GEMMs (Q + 4K + 4V), each 16x128x128 => 8 N-tile jobs apiece
  for (int job = wv; job < 72; job += 4) {
    const int task = job >> 3, nt = job & 7;
    int colbase, srcw;
    bf16_t* dst;
    if (task == 0)      { colbase = 0;   srcw = n;                        dst = sQ; }
    else if (task <= 4) { const int j = task - 1; colbase = 128;
                          srcw = ctx_idx[n * KCTX + j];                   dst = sK + j * 16 * EMB; }
    else                { const int j = task - 5; colbase = 256;
                          srcw = ctx_idx[n * KCTX + j];                   dst = sV + j * 16 * EMB; }
    const bf16_t* Abase = xe_bf + ((size_t)bbase * NWIN + srcw) * EMB;
    v8f c = {};
#pragma unroll
    for (int kc = 0; kc < 4; ++kc)
      c = wmma_bf(load_A16(Abase + kc * 32, ldXe),
                  load_B16(WT + (kc * 32) * 384 + colbase + nt * 16, 384), c);
    const float bias = ca_bqkv[n * 384 + colbase + nt * 16 + c_col()];
#pragma unroll
    for (int v = 0; v < 8; ++v) c[v] += bias;
    store_C_bf16(dst + nt * 16, EMB, c);
  }
  __syncthreads();

  // masked 4-way attention, one (b,h) per thread
  if (threadIdx.x < 64) {
    const int bi = threadIdx.x >> 2, h = threadIdx.x & 3;
    const float scale = 0.17677669529663687f;
    float q[32];
#pragma unroll
    for (int d = 0; d < 32; ++d) q[d] = (float)sQ[bi * EMB + h * 32 + d];
    float s[KCTX];
#pragma unroll
    for (int j = 0; j < KCTX; ++j) {
      if (ctx_mask[n * KCTX + j]) {
        float acc = 0.f;
#pragma unroll
        for (int d = 0; d < 32; ++d)
          acc += q[d] * (float)sK[j * 16 * EMB + bi * EMB + h * 32 + d];
        s[j] = acc * scale;
      } else {
        s[j] = -1e9f;
      }
    }
    float mx = fmaxf(fmaxf(s[0], s[1]), fmaxf(s[2], s[3]));
    float sum = 0.f;
#pragma unroll
    for (int j = 0; j < KCTX; ++j) { s[j] = __expf(s[j] - mx); sum += s[j]; }
    const float inv = fast_rcp(sum);
#pragma unroll
    for (int d = 0; d < 32; ++d) {
      float o = 0.f;
#pragma unroll
      for (int j = 0; j < KCTX; ++j)
        o += s[j] * (float)sV[j * 16 * EMB + bi * EMB + h * 32 + d];
      sO[bi * EMB + h * 32 + d] = (bf16_t)(o * inv);
    }
  }
  __syncthreads();

  // out = o @ ca_Wo^T + bo + residual(xe)
  const bf16_t* WoT = caWoT + (size_t)n * EMB * EMB;
  for (int nt = wv; nt < 8; nt += 4) {
    v8f c = {};
#pragma unroll
    for (int kc = 0; kc < 4; ++kc)
      c = wmma_bf(load_A16(sO + kc * 32, EMB),
                  load_B16(WoT + (kc * 32) * EMB + nt * 16, EMB), c);
    const int col = nt * 16 + c_col();
#pragma unroll
    for (int v = 0; v < 8; ++v) {
      const int row = c_row0() + v;
      c[v] += ca_bo[n * EMB + col] +
              xe_f[((size_t)(bbase + row) * NWIN + n) * EMB + col];
    }
    store_C_f32(sY + nt * 16, EMB, c);
  }
  __syncthreads();

  // per-row LayerNorm with per-window gamma/beta -> seq_bf
  if (threadIdx.x < 16) {
    const int r = threadIdx.x;
    float m = 0.f;
    for (int e = 0; e < EMB; ++e) m += sY[r * EMB + e];
    m *= (1.f / EMB);
    float var = 0.f;
    for (int e = 0; e < EMB; ++e) { const float d = sY[r * EMB + e] - m; var += d * d; }
    var *= (1.f / EMB);
    const float rinv = fast_rsqrt(var + 1e-5f);
    for (int e = 0; e < EMB; ++e) {
      const float val = (sY[r * EMB + e] - m) * rinv * ca_gamma[n * EMB + e] + ca_beta[n * EMB + e];
      seq_bf[((size_t)(bbase + r) * NWIN + n) * EMB + e] = (bf16_t)val;
    }
  }
}

// ---------------------------------------------------------------------------
// K4: MLP head.  64x16384x64 WMMA GEMM + ReLU, then tiny 64x5 output.
// ---------------------------------------------------------------------------
__global__ __launch_bounds__(256) void mlp_kernel(
    const bf16_t* __restrict__ seq_bf,   // [64][16384] row-major
    const bf16_t* __restrict__ W1T,      // [16384][64]
    const float* __restrict__ b1, const float* __restrict__ W2,
    const float* __restrict__ b2, float* __restrict__ out) {
  __shared__ float sH1[BATCH * 64];
  const int wv = threadIdx.x >> 5;
  const int ldA = NWIN * EMB;   // 16384

  for (int job = wv; job < 16; job += 8) {
    const int mi = job >> 2, ni = job & 3;
    v8f c = {};
    for (int kc = 0; kc < 512; ++kc)
      c = wmma_bf(load_A16(seq_bf + (size_t)(mi * 16) * ldA + (size_t)kc * 32, ldA),
                  load_B16(W1T + (size_t)(kc * 32) * 64 + ni * 16, 64), c);
    const int col = ni * 16 + c_col();
#pragma unroll
    for (int v = 0; v < 8; ++v) {
      const float t = c[v] + b1[col];
      c[v] = t > 0.f ? t : 0.f;
    }
    store_C_f32(sH1 + mi * 16 * 64 + ni * 16, 64, c);
  }
  __syncthreads();

  for (int i = threadIdx.x; i < BATCH * OUTD; i += 256) {
    const int b = i / OUTD, o = i % OUTD;
    float acc = b2[o];
    for (int u = 0; u < 64; ++u) acc += sH1[b * 64 + u] * W2[o * 64 + u];
    out[i] = acc;
  }
}

// ---------------------------------------------------------------------------
// Host-side orchestration
// ---------------------------------------------------------------------------
extern "C" void kernel_launch(void* const* d_in, const int* in_sizes, int n_in,
                              void* d_out, int out_size, void* d_ws, size_t ws_size,
                              hipStream_t stream) {
  (void)in_sizes; (void)n_in; (void)out_size; (void)ws_size;
  const float* x        = (const float*)d_in[0];
  const float* Wih_f    = (const float*)d_in[1];
  const float* Whh_f    = (const float*)d_in[2];
  const float* bih_f    = (const float*)d_in[3];
  const float* bhh_f    = (const float*)d_in[4];
  const float* Wih_b    = (const float*)d_in[5];
  const float* Whh_b    = (const float*)d_in[6];
  const float* bih_b    = (const float*)d_in[7];
  const float* bhh_b    = (const float*)d_in[8];
  const float* sa_Wqkv  = (const float*)d_in[9];
  const float* sa_bqkv  = (const float*)d_in[10];
  const float* sa_Wo    = (const float*)d_in[11];
  const float* sa_bo    = (const float*)d_in[12];
  const float* sa_gamma = (const float*)d_in[13];
  const float* sa_beta  = (const float*)d_in[14];
  const float* proj_W   = (const float*)d_in[15];
  const float* proj_b   = (const float*)d_in[16];
  const float* ca_Wqkv  = (const float*)d_in[17];
  const float* ca_bqkv  = (const float*)d_in[18];
  const float* ca_Wo    = (const float*)d_in[19];
  const float* ca_bo    = (const float*)d_in[20];
  const float* ca_gamma = (const float*)d_in[21];
  const float* ca_beta  = (const float*)d_in[22];
  const float* W1       = (const float*)d_in[23];
  const float* b1       = (const float*)d_in[24];
  const float* W2       = (const float*)d_in[25];
  const float* b2       = (const float*)d_in[26];
  const int* ctx_idx    = (const int*)d_in[27];
  const unsigned char* ctx_mask = (const unsigned char*)d_in[28];
  float* out = (float*)d_out;

  // workspace carve-up
  char* p = (char*)d_ws;
  auto carve = [&](size_t bytes) -> void* {
    void* r = (void*)p;
    p += (bytes + 255) & ~(size_t)255;
    return r;
  };
  bf16_t* h_all    = (bf16_t*)carve((size_t)NWINDOWS * WINLEN * EMB * sizeof(bf16_t));
  bf16_t* wWihT_f  = (bf16_t*)carve((size_t)32 * 256 * sizeof(bf16_t));
  bf16_t* wWihT_b  = (bf16_t*)carve((size_t)32 * 256 * sizeof(bf16_t));
  bf16_t* wWhhT_f  = (bf16_t*)carve((size_t)64 * 256 * sizeof(bf16_t));
  bf16_t* wWhhT_b  = (bf16_t*)carve((size_t)64 * 256 * sizeof(bf16_t));
  bf16_t* wQkvT    = (bf16_t*)carve((size_t)128 * 384 * sizeof(bf16_t));
  bf16_t* wWoT     = (bf16_t*)carve((size_t)128 * 128 * sizeof(bf16_t));
  bf16_t* wProjT   = (bf16_t*)carve((size_t)128 * 128 * sizeof(bf16_t));
  bf16_t* wCaQkvT  = (bf16_t*)carve((size_t)NWIN * 128 * 384 * sizeof(bf16_t));
  bf16_t* wCaWoT   = (bf16_t*)carve((size_t)NWIN * 128 * 128 * sizeof(bf16_t));
  bf16_t* wW1T     = (bf16_t*)carve((size_t)16384 * 64 * sizeof(bf16_t));
  bf16_t* xe_bf    = (bf16_t*)carve((size_t)BATCH * NWIN * EMB * sizeof(bf16_t));
  bf16_t* seq_bf   = (bf16_t*)carve((size_t)BATCH * NWIN * EMB * sizeof(bf16_t));
  float*  xe_f     = (float*)carve((size_t)BATCH * NWIN * EMB * sizeof(float));

  // K0: weight transposition f32 -> bf16 (B-operand layouts)
  bt_kernel<<<64,   256, 0, stream>>>(Wih_f,   wWihT_f, 1, 256, 32);
  bt_kernel<<<64,   256, 0, stream>>>(Wih_b,   wWihT_b, 1, 256, 32);
  bt_kernel<<<64,   256, 0, stream>>>(Whh_f,   wWhhT_f, 1, 256, 64);
  bt_kernel<<<64,   256, 0, stream>>>(Whh_b,   wWhhT_b, 1, 256, 64);
  bt_kernel<<<192,  256, 0, stream>>>(sa_Wqkv, wQkvT,   1, 384, 128);
  bt_kernel<<<64,   256, 0, stream>>>(sa_Wo,   wWoT,    1, 128, 128);
  bt_kernel<<<64,   256, 0, stream>>>(proj_W,  wProjT,  1, 128, 128);
  bt_kernel<<<2048, 256, 0, stream>>>(ca_Wqkv, wCaQkvT, NWIN, 384, 128);
  bt_kernel<<<1024, 256, 0, stream>>>(ca_Wo,   wCaWoT,  NWIN, 128, 128);
  bt_kernel<<<1024, 256, 0, stream>>>(W1,      wW1T,    1, 64, 16384);

  // K1: bidirectional LSTM scan (8192 windows x 2 directions)
  lstm_kernel<<<dim3(NWINDOWS / 16, 2), 256, 0, stream>>>(
      x, wWihT_f, wWhhT_f, bih_f, bhh_f, wWihT_b, wWhhT_b, bih_b, bhh_b, h_all);

  // K2: windowed self-attention + LN + proj + mean -> xe
  attn_kernel<<<NWINDOWS, 32, 0, stream>>>(
      h_all, wQkvT, sa_bqkv, wWoT, sa_bo, sa_gamma, sa_beta, wProjT, proj_b,
      xe_f, xe_bf);

  // K3: contextual cross-attention + LN -> seq
  cross_kernel<<<dim3(NWIN, BATCH / 16), 128, 0, stream>>>(
      xe_bf, xe_f, wCaQkvT, ca_bqkv, wCaWoT, ca_bo, ca_gamma, ca_beta,
      ctx_idx, ctx_mask, seq_bf);

  // K4: MLP head -> out
  mlp_kernel<<<1, 256, 0, stream>>>(seq_bf, wW1T, b1, W2, b2, out);
}
